// LSTMTemperatureRegressorV2_33586644254930
// MI455X (gfx1250) — matrix operand order repaired and checked by reading, not compiled, not run on hardware
//
#include <hip/hip_runtime.h>
#include <hip/hip_bf16.h>

#define B_     256
#define T_     200
#define FD     16
#define H_     256
#define FWD_   64
#define KPAD   288      // 32 (padded input chunk) + 256 (h)
#define NTILES 64       // 4H/16
#define KCH    9        // KPAD/32
#define ROWW   144      // dwords per A-row in LDS (KPAD/2)

typedef __attribute__((ext_vector_type(16))) __bf16 v16bf;
typedef __attribute__((ext_vector_type(8)))  float  v8f;

union Frag32 { uint4 u[2]; v16bf bf; };

__device__ __forceinline__ unsigned short f2bf(float f) {
  union { float f; unsigned int u; } c; c.f = f;
  unsigned int u = c.u;
  unsigned int r = u + 0x7FFFu + ((u >> 16) & 1u);   // round-to-nearest-even
  return (unsigned short)(r >> 16);
}

__device__ __forceinline__ float sigm(float x) { return 1.f / (1.f + __expf(-x)); }

// K index -> source element of the concatenated gate weight [Wih(18) | pad | Whh(256)]
__device__ __forceinline__ float wg_elem(const float* __restrict__ Wih,
                                         const float* __restrict__ Whh,
                                         int col, int K) {
  if (K < 18)  return Wih[col * 18 + K];
  if (K >= 32) return Whh[col * 256 + (K - 32)];
  return 0.f;
}

// ---- one-time pack: gate weights into WMMA B-fragment order (bf16 pairs) ----
// word index = ((nt*KCH + kk)*32 + lane)*8 + v ; B layout: VGPR v<4 holds
// K = 8*half + 2v, +1; v>=4 holds K = 16 + 8*half + 2(v-4), +1.
__global__ void pack_wg_kernel(const float* __restrict__ Wih,
                               const float* __restrict__ Whh,
                               unsigned int* __restrict__ wpack) {
  int t = blockIdx.x * blockDim.x + threadIdx.x;
  if (t >= NTILES * KCH * 32 * 8) return;
  int v    = t & 7;
  int lane = (t >> 3) & 31;
  int rem  = t >> 8;                 // nt*KCH + kk
  int kk   = rem % KCH;
  int nt   = rem / KCH;
  int col  = nt * 16 + (lane & 15);
  int half = lane >> 4;
  int K    = kk * 32 + 8 * half + 2 * (v & 3) + 16 * (v >> 2);
  float w0 = wg_elem(Wih, Whh, col, K);
  float w1 = wg_elem(Wih, Whh, col, K + 1);
  wpack[t] = (unsigned int)f2bf(w0) | ((unsigned int)f2bf(w1) << 16);
}

// ---- one-time pack: W1 (64x256) into B-fragment order ----
__global__ void pack_w1_kernel(const float* __restrict__ W1,
                               unsigned int* __restrict__ w1pack) {
  int t = blockIdx.x * blockDim.x + threadIdx.x;
  if (t >= 4 * 8 * 32 * 8) return;
  int v    = t & 7;
  int lane = (t >> 3) & 31;
  int rem  = t >> 8;                 // nt*8 + kk
  int kk   = rem & 7;
  int nt   = rem >> 3;
  int col  = nt * 16 + (lane & 15);
  int half = lane >> 4;
  int K    = kk * 32 + 8 * half + 2 * (v & 3) + 16 * (v >> 2);
  float w0 = W1[col * 256 + K];
  float w1 = W1[col * 256 + K + 1];
  w1pack[t] = (unsigned int)f2bf(w0) | ((unsigned int)f2bf(w1) << 16);
}

// ---- persistent LSTM kernel: 1 block = 16 batch rows for all T steps ----
__global__ __launch_bounds__(256) void lstm_persist_kernel(
    const float* __restrict__ z,  const float* __restrict__ x,
    const float* __restrict__ h0,
    const float* __restrict__ b,  const float* __restrict__ b1,
    const float* __restrict__ W2, const float* __restrict__ b2,
    const uint4* __restrict__ wpack, const uint4* __restrict__ w1pack,
    float* __restrict__ out)
{
  __shared__ unsigned short A2s[16 * KPAD];  // A matrix rows, bf16 (inp | h)
  __shared__ float hid_lds[FWD_ * 16];       // FFN hidden, [col][m] (conflict-free)
  __shared__ float b_lds[4 * H_];
  __shared__ float b1_lds[FWD_];
  __shared__ float W2_lds[FWD_];
  __shared__ float zp_lds[16];
  __shared__ float b2_s;

  const int tid   = threadIdx.x;
  const int wid   = tid >> 5;      // wave 0..7
  const int lane  = tid & 31;
  const int half  = lane >> 4;
  const int bbase = blockIdx.x * 16;
  unsigned int* A2w = (unsigned int*)A2s;

  // ---- preload constants & initial state ----
  for (int i = tid; i < 4 * H_; i += 256) b_lds[i] = b[i];
  if (tid < FWD_) { b1_lds[tid] = b1[tid]; W2_lds[tid] = W2[tid]; }
  if (tid == 0) b2_s = b2[0];
  if (tid < 16) zp_lds[tid] = 0.f;
  for (int i = tid; i < 16 * H_; i += 256) {
    int m = i >> 8, j = i & 255;
    A2s[m * KPAD + 32 + j] = f2bf(h0[(bbase + m) * H_ + j]);
  }
  float creg[2][8];                          // persistent cell state (f32, in VGPRs)
  #pragma unroll
  for (int G = 0; G < 2; G++)
    #pragma unroll
    for (int r = 0; r < 8; r++) creg[G][r] = 0.f;
  __syncthreads();

  const int arow = (lane & 15) * ROWW;       // A-frag row base (dwords)

  for (int t = 0; t < T_; t++) {
    // ---- Phase A: assemble input chunk K=0..31 (z, x, z-feedback, pad) ----
    {
      int m  = tid >> 4;
      int kp = tid & 15;
      float vals[2];
      #pragma unroll
      for (int q = 0; q < 2; q++) {
        int K = 2 * kp + q;
        float vv = 0.f;
        if (K == 0)       vv = z[(bbase + m) * T_ + t];
        else if (K <= 16) vv = x[((bbase + m) * T_ + t) * FD + (K - 1)];
        else if (K == 17) vv = zp_lds[m];
        vals[q] = vv;
      }
      A2w[m * ROWW + kp] =
          (unsigned int)f2bf(vals[0]) | ((unsigned int)f2bf(vals[1]) << 16);
    }
    __syncthreads();

    // ---- Phase B: g-tiles via WMMA; wave w owns N-tiles == w (mod 8) ----
    v8f acc[8];
    #pragma unroll
    for (int a = 0; a < 8; a++)
      #pragma unroll
      for (int r = 0; r < 8; r++) acc[a][r] = 0.f;
    for (int kk = 0; kk < KCH; kk++) {
      // Opaque per-lane BYTE OFFSET of this k-chunk's fragment base:
      //  - loads can't be proven loop-invariant -> no LICM hoist / scratch spill
      //  - pointer provenance stays the kernarg -> global_load (LOADcnt only)
      //  - all 8 fragments share ONE base + immediate a*73728 -> single 16-load
      //    clause, partial s_wait_loadcnt before each WMMA
      unsigned int off = ((unsigned)(wid * KCH + kk) * 32 + lane) * 32;
      asm volatile("" : "+v"(off));
      const char* wbase = (const char*)wpack + off;

      Frag32 fa;
      int base0 = kk * 16 + half * 4;   // A layout: K 8h..8h+7 then 16+8h..23+8h
      fa.u[0] = *(const uint4*)(A2w + arow + base0);
      fa.u[1] = *(const uint4*)(A2w + arow + base0 + 8);

      Frag32 fb[8];
      #pragma unroll
      for (int a = 0; a < 8; a++) {
        const uint4* bp = (const uint4*)(wbase + (size_t)a * (8 * KCH * 1024));
        fb[a].u[0] = bp[0];
        fb[a].u[1] = bp[1];
      }
      #pragma unroll
      for (int a = 0; a < 8; a++) {
        acc[a] = __builtin_amdgcn_wmma_f32_16x16x32_bf16(
            false, fa.bf, false, fb[a].bf, (short)0, acc[a], false, false);
      }
    }
    __syncthreads();

    // ---- Phase C: LSTM pointwise, all in registers; h -> A2 (bf16) ----
    // acc[2g+G] = gate g tile for unit group G (j/16 = wid + 8G)
    #pragma unroll
    for (int G = 0; G < 2; G++) {
      int j = (wid + 8 * G) * 16 + (lane & 15);
      float bi = b_lds[j], bff = b_lds[256 + j], bg = b_lds[512 + j], bo = b_lds[768 + j];
      #pragma unroll
      for (int r = 0; r < 8; r++) {
        int m = r + 8 * half;
        float gi = sigm (acc[0 + G][r] + bi);
        float gf = sigm (acc[2 + G][r] + bff);
        float gg = tanhf(acc[4 + G][r] + bg);
        float go = sigm (acc[6 + G][r] + bo);
        float c  = gf * creg[G][r] + gi * gg;
        creg[G][r] = c;
        float h  = go * tanhf(c);
        A2s[m * KPAD + 32 + j] = f2bf(h);
        if (t == T_ - 1) {
          out[B_ * T_ + (bbase + m) * H_ + j]            = h;  // hn
          out[B_ * T_ + B_ * H_ + (bbase + m) * H_ + j]  = c;  // cn
        }
      }
    }
    __syncthreads();

    // ---- Phase D: hid = relu(h @ W1^T + b1), 4 N-tiles on waves 0..3 ----
    if (wid < 4) {
      // One opaque base; the 8 k-chunk fragments sit at immediates kk*1024.
      unsigned int offd = ((unsigned)(wid * 8) * 32 + lane) * 32;
      asm volatile("" : "+v"(offd));
      const char* dbase = (const char*)w1pack + offd;

      v8f fac;
      #pragma unroll
      for (int r = 0; r < 8; r++) fac[r] = 0.f;
      #pragma unroll
      for (int kk = 0; kk < 8; kk++) {
        Frag32 fa;
        int base0 = (kk + 1) * 16 + half * 4;  // h region chunks
        fa.u[0] = *(const uint4*)(A2w + arow + base0);
        fa.u[1] = *(const uint4*)(A2w + arow + base0 + 8);
        const uint4* bp = (const uint4*)(dbase + kk * 1024);
        Frag32 fb;
        fb.u[0] = bp[0];
        fb.u[1] = bp[1];
        fac = __builtin_amdgcn_wmma_f32_16x16x32_bf16(
            false, fa.bf, false, fb.bf, (short)0, fac, false, false);
      }
      int col = wid * 16 + (lane & 15);
      float bb = b1_lds[col];
      #pragma unroll
      for (int r = 0; r < 8; r++) {
        int m = r + 8 * half;
        float v = fac[r] + bb;
        hid_lds[col * 16 + m] = v > 0.f ? v : 0.f;
      }
    }
    __syncthreads();

    // ---- Phase E: delta = relu(hid @ W2 + b2); monotone z feedback ----
    // One full wave: 2 lanes per batch row, 32-elem partial sums + shfl_xor(16).
    if (wid == 0) {
      int m = lane & 15;
      int p = lane >> 4;
      float s = 0.f;
      #pragma unroll 8
      for (int k = p * 32; k < p * 32 + 32; k++)
        s += hid_lds[k * 16 + m] * W2_lds[k];
      s += __shfl_xor(s, 16, 32);
      if (p == 0) {
        float sv = s + b2_s;
        float d  = sv > 0.f ? sv : 0.f;
        float zn = zp_lds[m] + d;
        zp_lds[m] = zn;
        out[(bbase + m) * T_ + t] = zn;
      }
    }
    __syncthreads();
  }
}

extern "C" void kernel_launch(void* const* d_in, const int* in_sizes, int n_in,
                              void* d_out, int out_size, void* d_ws, size_t ws_size,
                              hipStream_t stream) {
  const float* z   = (const float*)d_in[0];
  const float* x   = (const float*)d_in[1];
  const float* h0  = (const float*)d_in[2];
  const float* Wih = (const float*)d_in[3];
  const float* Whh = (const float*)d_in[4];
  const float* b   = (const float*)d_in[5];
  const float* W1  = (const float*)d_in[6];
  const float* b1  = (const float*)d_in[7];
  const float* W2  = (const float*)d_in[8];
  const float* b2  = (const float*)d_in[9];
  float* out = (float*)d_out;

  unsigned int* wpack  = (unsigned int*)d_ws;
  unsigned int* w1pack = wpack + NTILES * KCH * 32 * 8;   // 147456 dwords

  pack_wg_kernel<<<(NTILES * KCH * 32 * 8 + 255) / 256, 256, 0, stream>>>(Wih, Whh, wpack);
  pack_w1_kernel<<<(4 * 8 * 32 * 8 + 255) / 256, 256, 0, stream>>>(W1, w1pack);
  lstm_persist_kernel<<<B_ / 16, 256, 0, stream>>>(
      z, x, h0, b, b1, W2, b2,
      (const uint4*)wpack, (const uint4*)w1pack, out);
}
